// UNetSpherical_68564857913653
// MI455X (gfx1250) — matrix operand order repaired and checked by reading, not compiled, not run on hardware
//
#include <hip/hip_runtime.h>
#include <hip/hip_bf16.h>
#include <stddef.h>

// ---------------------------------------------------------------------------
// CDNA5 / gfx1250 spherical Chebyshev U-Net.
//  * SpMM: fp32, LDS-staged row structure.
//  * Channel-mixing GEMMs: v_wmma_f32_16x16x32_bf16, fp32 accumulate.
//    - W pre-packed once per GEMM into fragment-ordered bf16 (contiguous
//      32B B-fragment loads, no strided/scalar loads in the hot loop).
//    - Segment-hoisted K loop (no integer division in the hot loop).
//    - Template-selected 1 or 2 N-tiles per wave: no EXEC manipulation in
//      the hot loop; A fragment reused by two independent WMMAs.
//  * BN two-pass (LDS atomics -> global atomics), apply fused with ReLU +
//    residual add. Pool/argmax; one-hot unpool written into concat buffer.
// ---------------------------------------------------------------------------

#define BATCH 4
#define V0C   12288
#define V1C   (V0C/4)
#define V2C   (V0C/16)
#define M0C   (BATCH*V0C)
#define M1C   (BATCH*V1C)
#define M2C   (BATCH*V2C)
#define KNN_  20

typedef __attribute__((ext_vector_type(16))) __bf16 v16bf;
typedef __attribute__((ext_vector_type(8)))  float  v8f;

// ---------------------------------------------------------------------------
// SpMM:  y[b,v,f] = scale * (L x)[b,v,f] + addc * xadd[b,v,f]
// ---------------------------------------------------------------------------
__global__ void spmm_kernel(const float* __restrict__ xin,
                            const int*   __restrict__ cols,
                            const float* __restrict__ vals,
                            float*       __restrict__ y,
                            int V, int F, float scale,
                            const float* __restrict__ xadd, float addc)
{
    __shared__ int   sc[KNN_];
    __shared__ float sv[KNN_ + 1];
    const int bv = blockIdx.x;
    const int v  = bv % V;
    const int b  = bv / V;
    if (threadIdx.x < KNN_) {
        sc[threadIdx.x] = cols[v * KNN_ + threadIdx.x];
        sv[threadIdx.x] = vals[v * KNN_ + threadIdx.x];
    }
    if (threadIdx.x == KNN_) sv[KNN_] = vals[(size_t)V * KNN_ + v];   // diagonal
    __syncthreads();

    const float* xb = xin + (size_t)b * V * F;
    const size_t rowoff = ((size_t)b * V + v) * F;
    for (int f = threadIdx.x; f < F; f += blockDim.x) {
        float acc = sv[KNN_] * xb[(size_t)v * F + f];
        #pragma unroll 5
        for (int j = 0; j < KNN_; ++j)
            acc += sv[j] * xb[(size_t)sc[j] * F + f];
        float r = scale * acc;
        if (xadd) r += addc * xadd[rowoff + f];
        y[rowoff + f] = r;
    }
}

// ---------------------------------------------------------------------------
// Pack W (fp32, (Kdim,fout) row-major or (fout,Kdim) row-major) into
// WMMA-B-fragment-ordered bf16:  wp[((kt*tilesN + tn)*32 + lane)*16 + e]
// where for lane/e:  K = kt*32 + (lane&16 ? 8:0) + (e<8 ? e : e+8),  n = tn*16 + (lane&15).
// ---------------------------------------------------------------------------
__global__ void pack_w_kernel(const float* __restrict__ w, __bf16* __restrict__ wp,
                              int kdim, int fout, int wtrans)
{
    const int t = blockIdx.x * blockDim.x + threadIdx.x;
    if (t >= kdim * fout) return;
    const int e    = t & 15;
    const int lane = (t >> 4) & 31;
    const int tile = t >> 9;                 // kt*tilesN + tn   (32*16 = 512)
    const int tilesN = fout >> 4;
    const int kt = tile / tilesN;
    const int tn = tile - kt * tilesN;
    const int K  = kt * 32 + ((lane & 16) ? 8 : 0) + (e < 8 ? e : e + 8);
    const int n  = tn * 16 + (lane & 15);
    const float v = wtrans ? w[(size_t)n * kdim + K] : w[(size_t)K * fout + n];
    wp[t] = (__bf16)v;
}

// ---------------------------------------------------------------------------
// bf16 WMMA GEMM:  out[m,n] = sum_k A[m,k] * W[k,n] + bias[n]
// A = virtual concat of up to 3 fp32 sources (x0|x1|x2), each (M, fin).
// W pre-packed by pack_w_kernel. TWO=true -> one wave computes a 16x32
// output (two N tiles, shared A fragment, independent accumulators).
// Requires M%16==0, fout%16==0, fin%32==0; TWO requires fout%32==0.
// ---------------------------------------------------------------------------
template <bool TWO>
__global__ void gemm_bf16_wmma(const float* __restrict__ x0,
                               const float* __restrict__ x1,
                               const float* __restrict__ x2,
                               const __bf16* __restrict__ wp,
                               const float* __restrict__ bias,
                               float*       __restrict__ out,
                               int M, int fin, int fout,
                               int nsrc, int totalUnits)
{
    const int wave = threadIdx.x >> 5;
    const int lane = threadIdx.x & 31;
    const int unit = blockIdx.x * 8 + wave;
    if (unit >= totalUnits) return;            // wave-uniform: EXEC all-1 at WMMA

    const int tilesN = fout >> 4;
    const int unitsN = TWO ? (tilesN >> 1) : tilesN;
    const int tm  = unit / unitsN;
    const int tp  = unit - tm * unitsN;
    const int tn0 = TWO ? tp * 2 : tp;

    const int row = tm * 16 + (lane & 15);
    const int hi  = (lane >> 4) & 1;
    const int kbA = hi * 8;

    // B-fragment stream: 32 contiguous bytes per lane per k-step.
    const __bf16* wr = wp + (size_t)tn0 * 512 + (size_t)lane * 16;
    const size_t wstep = (size_t)tilesN * 512;

    v8f acc0 = {0.f,0.f,0.f,0.f,0.f,0.f,0.f,0.f};
    v8f acc1 = {0.f,0.f,0.f,0.f,0.f,0.f,0.f,0.f};

    for (int s2 = 0; s2 < nsrc; ++s2) {
        const float* sp = (s2 == 0) ? x0 : ((s2 == 1) ? x1 : x2);
        const float* abase = sp + (size_t)row * fin + kbA;
        #pragma unroll 2
        for (int kf = 0; kf < fin; kf += 32) {
            const float4* p = (const float4*)(abase + kf);
            const float4 q0 = p[0];            // K = kf+kbA    .. +7
            const float4 q1 = p[1];
            const float4 q2 = p[4];            // K = kf+kbA+16 .. +23
            const float4 q3 = p[5];
            v16bf a;
            a[0]=(__bf16)q0.x; a[1]=(__bf16)q0.y; a[2]=(__bf16)q0.z; a[3]=(__bf16)q0.w;
            a[4]=(__bf16)q1.x; a[5]=(__bf16)q1.y; a[6]=(__bf16)q1.z; a[7]=(__bf16)q1.w;
            a[8]=(__bf16)q2.x; a[9]=(__bf16)q2.y; a[10]=(__bf16)q2.z; a[11]=(__bf16)q2.w;
            a[12]=(__bf16)q3.x; a[13]=(__bf16)q3.y; a[14]=(__bf16)q3.z; a[15]=(__bf16)q3.w;

            const v16bf b0 = *(const v16bf*)wr;
            acc0 = __builtin_amdgcn_wmma_f32_16x16x32_bf16(
                       false, a, false, b0, (short)0, acc0, false, false);
            if constexpr (TWO) {
                const v16bf b1 = *(const v16bf*)(wr + 512);
                acc1 = __builtin_amdgcn_wmma_f32_16x16x32_bf16(
                           false, a, false, b1, (short)0, acc1, false, false);
            }
            wr += wstep;
        }
    }

    const int nc0 = tn0 * 16 + (lane & 15);
    const float b0v = bias ? bias[nc0] : 0.f;
    #pragma unroll
    for (int j = 0; j < 8; ++j) {              // C/D layout: VGPR j -> M = j + 8*hi
        const int r = tm * 16 + j + hi * 8;
        out[(size_t)r * fout + nc0] = acc0[j] + b0v;
    }
    if constexpr (TWO) {
        const int nc1 = nc0 + 16;
        const float b1v = bias ? bias[nc1] : 0.f;
        #pragma unroll
        for (int j = 0; j < 8; ++j) {
            const int r = tm * 16 + j + hi * 8;
            out[(size_t)r * fout + nc1] = acc1[j] + b1v;
        }
    }
}

// ---------------------------------------------------------------------------
// BatchNorm helpers
// ---------------------------------------------------------------------------
__global__ void zero_kernel(float* __restrict__ p, int n)
{
    int i = blockIdx.x * blockDim.x + threadIdx.x;
    if (i < n) p[i] = 0.f;
}

__global__ void bn_reduce_kernel(const float* __restrict__ x, int Mrows, int fout,
                                 float* __restrict__ stats)   // [0:fout)=sum, [fout:)=sumsq
{
    __shared__ float s[1024];                                  // fout <= 512
    for (int i = threadIdx.x; i < 2 * fout; i += blockDim.x) s[i] = 0.f;
    __syncthreads();
    const int total = Mrows * fout;
    for (int i = blockIdx.x * blockDim.x + threadIdx.x; i < total;
         i += gridDim.x * blockDim.x) {
        const float v = x[i];
        const int   c = i % fout;
        atomicAdd(&s[c], v);
        atomicAdd(&s[fout + c], v * v);
    }
    __syncthreads();
    for (int i = threadIdx.x; i < 2 * fout; i += blockDim.x)
        atomicAdd(&stats[i], s[i]);
}

__global__ void bn_apply_kernel(const float* __restrict__ x,
                                const float* __restrict__ stats,
                                const float* __restrict__ g,
                                const float* __restrict__ beta,
                                int relu, const float* __restrict__ add,
                                float* __restrict__ out, int Mrows, int fout)
{
    const int i = blockIdx.x * blockDim.x + threadIdx.x;
    if (i >= Mrows * fout) return;
    const int   c   = i % fout;
    const float inv = 1.0f / (float)Mrows;
    const float m   = stats[c] * inv;
    const float var = stats[fout + c] * inv - m * m;
    const float rs  = __frsqrt_rn(var + 1e-5f);
    float yv = g[c] * (x[i] - m) * rs + beta[c];
    if (relu) yv = fmaxf(yv, 0.f);
    if (add)  yv += add[i];
    out[i] = yv;
}

// ---------------------------------------------------------------------------
// Pool (max over 4 vertices, first-max argmax) / unpool / strided copy
// ---------------------------------------------------------------------------
__global__ void pool_kernel(const float* __restrict__ x, float* __restrict__ out,
                            int* __restrict__ idx, int MpRows, int F)
{
    const int t = blockIdx.x * blockDim.x + threadIdx.x;
    if (t >= MpRows * F) return;
    const int row = t / F, f = t % F;
    const size_t base = ((size_t)row * 4) * F + f;
    float best = x[base]; int bi = 0;
    #pragma unroll
    for (int j = 1; j < 4; ++j) {
        const float v = x[base + (size_t)j * F];
        if (v > best) { best = v; bi = j; }
    }
    out[t] = best;
    idx[t] = bi;
}

__global__ void unpool_kernel(const float* __restrict__ x, const int* __restrict__ idx,
                              float* __restrict__ dst, int MpRows, int F,
                              int Ftot, int coff)
{
    const int t = blockIdx.x * blockDim.x + threadIdx.x;
    if (t >= MpRows * F) return;
    const int row = t / F, f = t % F;
    const int j = idx[t];
    const float v = x[t];
    #pragma unroll
    for (int jj = 0; jj < 4; ++jj)
        dst[((size_t)(row * 4 + jj)) * Ftot + coff + f] = (jj == j) ? v : 0.f;
}

__global__ void copy_strided_kernel(const float* __restrict__ src, float* __restrict__ dst,
                                    int Mrows, int F, int Ftot, int coff)
{
    const int t = blockIdx.x * blockDim.x + threadIdx.x;
    if (t >= Mrows * F) return;
    const int row = t / F, f = t % F;
    dst[(size_t)row * Ftot + coff + f] = src[t];
}

// ---------------------------------------------------------------------------
// Host-side helpers
// ---------------------------------------------------------------------------
static void run_spmm(const float* xin, const int* cols, const float* vals, float* y,
                     int V, int F, float scale, const float* xadd, float addc,
                     hipStream_t s)
{
    spmm_kernel<<<dim3(BATCH * V), dim3(128), 0, s>>>(xin, cols, vals, y, V, F,
                                                      scale, xadd, addc);
}

static void run_gemm(const float* x0, const float* x1, const float* x2,
                     const float* w, const float* bias, float* out, __bf16* wp,
                     int M, int fin, int fout, int nsrc, int wtrans, hipStream_t s)
{
    const int kdim = nsrc * fin;
    const int wtot = kdim * fout;
    pack_w_kernel<<<dim3((wtot + 255) / 256), dim3(256), 0, s>>>(w, wp, kdim, fout, wtrans);
    const int tilesN = fout / 16;
    if ((fout & 31) == 0) {
        const int units = (M / 16) * (tilesN / 2);
        gemm_bf16_wmma<true><<<dim3((units + 7) / 8), dim3(256), 0, s>>>(
            x0, x1, x2, wp, bias, out, M, fin, fout, nsrc, units);
    } else {
        const int units = (M / 16) * tilesN;
        gemm_bf16_wmma<false><<<dim3((units + 7) / 8), dim3(256), 0, s>>>(
            x0, x1, x2, wp, bias, out, M, fin, fout, nsrc, units);
    }
}

static void run_bn(const float* t, int M, int fout, const float* g, const float* be,
                   int relu, const float* add, float* out, float* stats, hipStream_t s)
{
    zero_kernel<<<dim3(4), dim3(256), 0, s>>>(stats, 2 * fout);
    bn_reduce_kernel<<<dim3(240), dim3(256), 0, s>>>(t, M, fout, stats);
    const int n = M * fout;
    bn_apply_kernel<<<dim3((n + 255) / 256), dim3(256), 0, s>>>(
        t, stats, g, be, relu, add, out, M, fout);
}

static void run_cheb(const float* xin, int M, int V, int fin, int fout,
                     const int* cols, const float* vals,
                     const float* w, const float* bias, __bf16* wp,
                     float* x1, float* x2, float* tout, hipStream_t s)
{
    run_spmm(xin, cols, vals, x1, V, fin, 1.0f, nullptr, 0.f, s);
    run_spmm(x1, cols, vals, x2, V, fin, 2.0f, xin, -1.0f, s);
    run_gemm(xin, x1, x2, w, bias, tout, wp, M, fin, fout, 3, 0, s);
}

// ---------------------------------------------------------------------------
// Entry point
// ---------------------------------------------------------------------------
extern "C" void kernel_launch(void* const* d_in, const int* in_sizes, int n_in,
                              void* d_out, int out_size, void* d_ws, size_t ws_size,
                              hipStream_t stream)
{
    (void)in_sizes; (void)n_in; (void)out_size; (void)ws_size;

    // ---- inputs (setup_inputs dict order, recursively flattened) ----
    const float* x     = (const float*)d_in[0];
    const float* c11w  = (const float*)d_in[1],  *c11b = (const float*)d_in[2];
    const float* c11g  = (const float*)d_in[3],  *c11be= (const float*)d_in[4];
    const float* c13w  = (const float*)d_in[5],  *c13b = (const float*)d_in[6];
    const float* c13g  = (const float*)d_in[7],  *c13be= (const float*)d_in[8];
    const float* r1w   = (const float*)d_in[9],  *r1b  = (const float*)d_in[10];
    const float* c21w  = (const float*)d_in[11], *c21b = (const float*)d_in[12];
    const float* c21g  = (const float*)d_in[13], *c21be= (const float*)d_in[14];
    const float* c23w  = (const float*)d_in[15], *c23b = (const float*)d_in[16];
    const float* c23g  = (const float*)d_in[17], *c23be= (const float*)d_in[18];
    const float* r2w   = (const float*)d_in[19], *r2b  = (const float*)d_in[20];
    const float* c31w  = (const float*)d_in[21], *c31b = (const float*)d_in[22];
    const float* c31g  = (const float*)d_in[23], *c31be= (const float*)d_in[24];
    const float* c33w  = (const float*)d_in[25], *c33b = (const float*)d_in[26];
    const float* c33g  = (const float*)d_in[27], *c33be= (const float*)d_in[28];
    const float* r3w   = (const float*)d_in[29], *r3b  = (const float*)d_in[30];
    const float* u21w  = (const float*)d_in[31], *u21b = (const float*)d_in[32];
    const float* u21g  = (const float*)d_in[33], *u21be= (const float*)d_in[34];
    const float* u22w  = (const float*)d_in[35], *u22b = (const float*)d_in[36];
    const float* u22g  = (const float*)d_in[37], *u22be= (const float*)d_in[38];
    const float* u11w  = (const float*)d_in[39], *u11b = (const float*)d_in[40];
    const float* u11g  = (const float*)d_in[41], *u11be= (const float*)d_in[42];
    const float* u12w  = (const float*)d_in[43], *u12b = (const float*)d_in[44];
    const float* u12g  = (const float*)d_in[45], *u12be= (const float*)d_in[46];
    const float* u13w  = (const float*)d_in[47], *u13b = (const float*)d_in[48];
    const int*   l0c   = (const int*)  d_in[50]; const float* l0v = (const float*)d_in[51];
    const int*   l1c   = (const int*)  d_in[54]; const float* l1v = (const float*)d_in[55];
    const int*   l2c   = (const int*)  d_in[58]; const float* l2v = (const float*)d_in[59];

    float* yout = (float*)d_out;

    // ---- workspace arenas (fp32 elements; 256B-aligned bumps) ----
    float* W = (float*)d_ws;
    size_t off = 0;
    auto arena = [&](size_t n) { size_t o = off; off += (n + 63) & ~(size_t)63; return W + o; };

    float*  aX1  = arena((size_t)M0C * 256);   // cheb x1 scratch
    float*  aX2  = arena((size_t)M0C * 256);   // cheb x2 scratch
    float*  aT   = arena((size_t)M0C * 128);   // GEMM out (pre-BN)
    float*  aR   = arena((size_t)M0C * 128);   // residual / ping buffer
    float*  aBN  = arena((size_t)M0C * 128);   // BN out / pong buffer
    float*  aCAT = arena((size_t)M0C * 256);   // concat / decoder input
    float*  e11  = arena((size_t)M0C * 64);    // skip: x_enc11
    float*  e1   = arena((size_t)M0C * 128);   // skip: x_enc1
    float*  x2i  = arena((size_t)M1C * 128);   // pooled level-1 input
    float*  e2   = arena((size_t)M1C * 256);   // skip: x_enc2
    float*  x3i  = arena((size_t)M2C * 256);   // pooled level-2 input
    int*    idx1 = (int*)arena((size_t)M1C * 128);
    int*    idx2 = (int*)arena((size_t)M2C * 256);
    float*  stats= arena(1024);
    __bf16* wpk  = (__bf16*)arena(262144);     // packed-weight buffer (<= 393216 bf16)

    const int T256 = 256;
    hipStream_t s = stream;

    // ================= Encoder, level 0 (V=12288) =================
    run_cheb(x, M0C, V0C, 32, 64, l0c, l0v, c11w, c11b, wpk, aX1, aX2, aT, s);
    run_bn(aT, M0C, 64, c11g, c11be, 1, nullptr, e11, stats, s);
    run_cheb(e11, M0C, V0C, 64, 128, l0c, l0v, c13w, c13b, wpk, aX1, aX2, aT, s);
    run_gemm(x, nullptr, nullptr, r1w, r1b, aR, wpk, M0C, 32, 128, 1, 1, s);
    run_bn(aT, M0C, 128, c13g, c13be, 1, aR, e1, stats, s);
    {
        int n = M1C * 128;
        pool_kernel<<<dim3((n + T256 - 1) / T256), dim3(T256), 0, s>>>(e1, x2i, idx1, M1C, 128);
    }

    // ================= Encoder, level 1 (V=3072) =================
    run_cheb(x2i, M1C, V1C, 128, 192, l1c, l1v, c21w, c21b, wpk, aX1, aX2, aT, s);
    run_bn(aT, M1C, 192, c21g, c21be, 1, nullptr, aBN, stats, s);
    run_cheb(aBN, M1C, V1C, 192, 256, l1c, l1v, c23w, c23b, wpk, aX1, aX2, aT, s);
    run_gemm(x2i, nullptr, nullptr, r2w, r2b, aR, wpk, M1C, 128, 256, 1, 1, s);
    run_bn(aT, M1C, 256, c23g, c23be, 1, aR, e2, stats, s);
    {
        int n = M2C * 256;
        pool_kernel<<<dim3((n + T256 - 1) / T256), dim3(T256), 0, s>>>(e2, x3i, idx2, M2C, 256);
    }

    // ================= Bottleneck, level 2 (V=768) =================
    run_cheb(x3i, M2C, V2C, 256, 512, l2c, l2v, c31w, c31b, wpk, aX1, aX2, aT, s);
    run_bn(aT, M2C, 512, c31g, c31be, 1, nullptr, aBN, stats, s);
    run_cheb(aBN, M2C, V2C, 512, 256, l2c, l2v, c33w, c33b, wpk, aX1, aX2, aT, s);
    run_gemm(x3i, nullptr, nullptr, r3w, r3b, aR, wpk, M2C, 256, 256, 1, 1, s);
    run_bn(aT, M2C, 256, c33g, c33be, 1, aR, aBN, stats, s);      // e3 -> aBN

    // ================= Decoder, level 1 =================
    {
        int n = M2C * 256;
        unpool_kernel<<<dim3((n + T256 - 1) / T256), dim3(T256), 0, s>>>(aBN, idx2, aCAT, M2C, 256, 512, 0);
        int m = M1C * 256;
        copy_strided_kernel<<<dim3((m + T256 - 1) / T256), dim3(T256), 0, s>>>(e2, aCAT, M1C, 256, 512, 256);
    }
    run_cheb(aCAT, M1C, V1C, 512, 256, l1c, l1v, u21w, u21b, wpk, aX1, aX2, aT, s);
    run_bn(aT, M1C, 256, u21g, u21be, 1, nullptr, aBN, stats, s);
    run_cheb(aBN, M1C, V1C, 256, 128, l1c, l1v, u22w, u22b, wpk, aX1, aX2, aT, s);
    run_bn(aT, M1C, 128, u22g, u22be, 1, nullptr, aR, stats, s);   // d128 -> aR

    // ================= Decoder, level 0 =================
    {
        int n = M1C * 128;
        unpool_kernel<<<dim3((n + T256 - 1) / T256), dim3(T256), 0, s>>>(aR, idx1, aCAT, M1C, 128, 256, 0);
        int m = M0C * 128;
        copy_strided_kernel<<<dim3((m + T256 - 1) / T256), dim3(T256), 0, s>>>(e1, aCAT, M0C, 128, 256, 128);
    }
    run_cheb(aCAT, M0C, V0C, 256, 128, l0c, l0v, u11w, u11b, wpk, aX1, aX2, aT, s);
    run_bn(aT, M0C, 128, u11g, u11be, 1, nullptr, aBN, stats, s);
    run_cheb(aBN, M0C, V0C, 128, 64, l0c, l0v, u12w, u12b, wpk, aX1, aX2, aT, s);
    run_bn(aT, M0C, 64, u12g, u12be, 1, nullptr, aR, stats, s);    // f64 -> aR

    // final concat [f64 | e11], then u13: 128 -> 16, bias fused, straight to d_out.
    {
        int n = M0C * 64;
        copy_strided_kernel<<<dim3((n + T256 - 1) / T256), dim3(T256), 0, s>>>(aR,  aCAT, M0C, 64, 128, 0);
        copy_strided_kernel<<<dim3((n + T256 - 1) / T256), dim3(T256), 0, s>>>(e11, aCAT, M0C, 64, 128, 64);
    }
    run_cheb(aCAT, M0C, V0C, 128, 16, l0c, l0v, u13w, u13b, wpk, aX1, aX2, yout, s);
}
